// GNN_Encoder_80410377715934
// MI455X (gfx1250) — compile-verified, matrix operand order checked by reading, not run on hardware
//
#include <hip/hip_runtime.h>
#include <hip/hip_bf16.h>
#include <stdint.h>

// ---------------------------------------------------------------------------
// GNN encoder for MI455X (gfx1250): bf16 WMMA GEMMs + fused hypernetwork
// with async global->LDS double-buffered weight panels.
// ---------------------------------------------------------------------------

typedef __attribute__((ext_vector_type(16))) __bf16 v16bf;
typedef __attribute__((ext_vector_type(8)))  __bf16 v8bf;
typedef __attribute__((ext_vector_type(8)))  float  v8f;

#define NFEAT 256
#define HID   64     // NHID
#define HID2  128    // 2*NHID
#define HID3  192    // 3*NHID
#define WMCOL 4096   // NHID*NHID

__device__ __forceinline__ __bf16 f2bf(float f) { return (__bf16)f; }

// CDNA5 async copies: global memory -> LDS, ASYNCcnt-tracked.
__device__ __forceinline__ void async_copy_b128(unsigned lds_off, const void* gptr) {
    asm volatile("global_load_async_to_lds_b128 %0, %1, off"
                 :: "v"(lds_off), "v"(gptr) : "memory");
}
__device__ __forceinline__ void async_copy_b64(unsigned lds_off, const void* gptr) {
    asm volatile("global_load_async_to_lds_b64 %0, %1, off"
                 :: "v"(lds_off), "v"(gptr) : "memory");
}
__device__ __forceinline__ void wait_async0() {
    asm volatile("s_wait_asynccnt 0" ::: "memory");
}

// XOR-butterfly add via ds_swizzle (group-of-32 mode: and=0x1f, or=0, xor=MASK)
template <int MASK>
__device__ __forceinline__ float xor_add(float x) {
    int y = __builtin_amdgcn_ds_swizzle(__float_as_int(x), (MASK << 10) | 0x1f);
    return x + __int_as_float(y);
}

// ------------------------- small elementwise kernels -----------------------

__global__ void zero_f32(float* p, long long n) {
    long long i = (long long)blockIdx.x * blockDim.x + threadIdx.x;
    if (i < n) p[i] = 0.0f;
}

__global__ void init_deg(float* deg, int n) {
    int i = blockIdx.x * blockDim.x + threadIdx.x;
    if (i < n) deg[i] = 1.0f;               // self-loop contributes 1
}

__global__ void accum_deg(const long long* ei, float* deg, int E) {
    int e = blockIdx.x * blockDim.x + threadIdx.x;
    if (e < E) atomicAdd(&deg[ei[(long long)E + e]], 1.0f);  // dst row
}

__global__ void make_dinv(float* deg, int n) {
    int i = blockIdx.x * blockDim.x + threadIdx.x;
    if (i < n) deg[i] = rsqrtf(deg[i]);     // deg >= 1 always
}

__global__ void cvt_f32_bf16(const float* in, __bf16* out, long long n) {
    long long i = (long long)blockIdx.x * blockDim.x + threadIdx.x;
    if (i < n) out[i] = f2bf(in[i]);
}

// ------------------------- B-matrix packing for WMMA -----------------------
// Packs a row-major f32 W[K x Nc] into per-lane-contiguous bf16 fragments:
// out[((colTile*ksteps + kk)*32 + lane)*16 + t], matching the 16-bit B layout
// (lane = N within tile; lanes 0-15 hold K {0..7,16..23}, 16-31 {8..15,24..31}).
__global__ void pack_b(const float* W, __bf16* out, int K, int Nc) {
    int ksteps = K >> 5;
    int id = blockIdx.x * blockDim.x + threadIdx.x;
    int total = (Nc >> 4) * ksteps * 32;
    if (id >= total) return;
    int lane = id & 31;
    int kk = (id >> 5) % ksteps;
    int ct = (id >> 5) / ksteps;
    int ncol = ct * 16 + (lane & 15);
    int off = (lane < 16) ? 0 : 8;
    int kbase = kk * 32 + off;
    __bf16* o = out + (long long)id * 16;
#pragma unroll
    for (int t = 0; t < 8; ++t) o[t]     = f2bf(W[(long long)(kbase + t)      * Nc + ncol]);
#pragma unroll
    for (int t = 0; t < 8; ++t) o[8 + t] = f2bf(W[(long long)(kbase + 16 + t) * Nc + ncol]);
}

// ------------------------------ generic GEMM -------------------------------
// C[nRows x Nc] = A_bf16[nRows x K] @ Bpacked (+bias), written as f32 (outF)
// and/or bf16 (outB) with row stride ldOut. One wave per 16x16 tile.
// KSTEPS = K/32 compile-time -> fully unrolled, constant instruction offsets.
template <int KSTEPS>
__global__ __launch_bounds__(128) void wmma_gemm(
    const __bf16* __restrict__ A, const __bf16* __restrict__ Bp,
    const float* __restrict__ bias,
    float* outF, __bf16* outB, int ldOut,
    int nRows, int Nc)
{
    int nct = Nc >> 4;
    int wave = blockIdx.x * (blockDim.x >> 5) + (threadIdx.x >> 5);
    int totalTiles = (nRows >> 4) * nct;
    if (wave >= totalTiles) return;
    int rg = wave / nct;
    int ct = wave % nct;
    int lane = threadIdx.x & 31;
    int ln = lane & 15;
    int row0 = rg << 4;
    int off = (lane < 16) ? 0 : 8;       // A K-offset and C row-half
    const __bf16* arow = A + (long long)(row0 + ln) * (KSTEPS * 32) + off;
    const __bf16* bp   = Bp + (long long)ct * (KSTEPS * 512) + (long long)lane * 16;

    v8f c = {};
#pragma unroll
    for (int kk = 0; kk < KSTEPS; ++kk) {
        v8bf lo = *(const v8bf*)(arow + kk * 32);
        v8bf hi = *(const v8bf*)(arow + kk * 32 + 16);
        v16bf a;
#pragma unroll
        for (int t = 0; t < 8; ++t) { a[t] = lo[t]; a[8 + t] = hi[t]; }
        v16bf b = *(const v16bf*)(bp + kk * 512);
        c = __builtin_amdgcn_wmma_f32_16x16x32_bf16(false, a, false, b,
                                                    (short)0, c, false, false);
    }
#pragma unroll
    for (int v = 0; v < 8; ++v) {
        int rr = row0 + off + v;
        int cc = ct * 16 + ln;
        float val = c[v] + (bias ? bias[cc] : 0.0f);
        long long o = (long long)rr * ldOut + cc;
        if (outF) outF[o] = val;
        if (outB) outB[o] = f2bf(val);
    }
}

// ------------------------------ edge scatter -------------------------------
// One wave per edge; lane handles F/32 contiguous floats; f32 global atomics.
__global__ __launch_bounds__(128) void scatter_edges(
    const long long* __restrict__ ei, const float* __restrict__ dinv,
    const float* __restrict__ hw, float* agg, int E, int F)
{
    int e = blockIdx.x * (blockDim.x >> 5) + (threadIdx.x >> 5);
    if (e >= E) return;
    int lane = threadIdx.x & 31;
    long long s = ei[e];
    long long d = ei[(long long)E + e];
    float nrm = dinv[s] * dinv[d];
    int per = F >> 5;
    const float* hs = hw + s * F + lane * per;
    float* ad = agg + d * F + lane * per;
#pragma unroll 4
    for (int t = 0; t < per; ++t)
        atomicAdd(&ad[t], nrm * hs[t]);
}

// h = relu(agg1 + dinv^2*hw1 + b1) -> bf16 (layer-1 epilogue, self-loop folded)
__global__ void finish1(const float* agg, const float* hw, const float* dinv,
                        const float* b1, __bf16* h_bf, int n) {
    long long i = (long long)blockIdx.x * blockDim.x + threadIdx.x;
    if (i >= (long long)n * HID2) return;
    int node = (int)(i >> 7), k = (int)(i & 127);
    float dv = dinv[node];
    float v = agg[i] + dv * dv * hw[i] + b1[k];
    h_bf[i] = f2bf(v > 0.0f ? v : 0.0f);
}

// res = agg2 + dinv^2*hw2 + b2 -> d_out[0..N*64); z[:,128..191] = bf16(relu(res))
__global__ void finish2(const float* agg, const float* hw, const float* dinv,
                        const float* b2, float* res_out, __bf16* z_bf, int n) {
    long long i = (long long)blockIdx.x * blockDim.x + threadIdx.x;
    if (i >= (long long)n * HID) return;
    int node = (int)(i >> 6), k = (int)(i & 63);
    float dv = dinv[node];
    float v = agg[i] + dv * dv * hw[i] + b2[k];
    res_out[i] = v;
    z_bf[(long long)node * HID3 + HID2 + k] = f2bf(v > 0.0f ? v : 0.0f);
}

// --------------------- fused hypernetwork (dominant GEMM) ------------------
// hid1[n,i] = sum_j (z[n]@Wm + bm)[i*64+j] * relu(res)[n,j]
// 8 waves/block, 16 nodes per wave. A fragments (z, K=192) resident in 48
// VGPRs; hr operands resident in 32 VGPRs (statically indexed). Wm column
// panels (6 KB) are double-buffered in LDS via CDNA5 async global->LDS
// copies (one b128 + one b64 per thread -> fully uniform, no exec masking),
// shared by all 8 waves. Inactive tail waves are clamped onto group 0
// (redundant compute, predicated stores) to keep EXEC uniform in the loop.
__global__ __launch_bounds__(256) void hyper_kernel(
    const __bf16* __restrict__ zb, const __bf16* __restrict__ wmp,
    const float* __restrict__ res, const float* __restrict__ bm,
    float* __restrict__ hid1, int n)
{
    __shared__ __bf16 bpanel[2][6 * 512];        // 2 x 6KB double buffer

    int wave = threadIdx.x >> 5;
    int lane = threadIdx.x & 31;
    int ln = lane & 15;
    int mboff = (lane < 16) ? 0 : 8;             // A K-offset == C row-half
    int group = blockIdx.x * 8 + wave;
    int ngroups = n >> 4;
    bool active = (group < ngroups);
    if (group >= ngroups) group = 0;             // clamp: uniform EXEC in loop
    int row0 = group << 4;

    // hr operands in registers: hr[row0+mboff+v][sub*16+ln] (static indexing)
    float hrreg[4][8];
#pragma unroll
    for (int sub = 0; sub < 4; ++sub)
#pragma unroll
        for (int v = 0; v < 8; ++v) {
            float t = res[(long long)(row0 + mboff + v) * HID + sub * 16 + ln];
            hrreg[sub][v] = t > 0.0f ? t : 0.0f;
        }

    // A fragments (z row for this lane, 6 k-steps)
    v16bf afrag[6];
    {
        const __bf16* zrow = zb + (long long)(row0 + ln) * HID3 + mboff;
#pragma unroll
        for (int kk = 0; kk < 6; ++kk) {
            v8bf lo = *(const v8bf*)(zrow + kk * 32);
            v8bf hi = *(const v8bf*)(zrow + kk * 32 + 16);
            v16bf a;
#pragma unroll
            for (int t = 0; t < 8; ++t) { a[t] = lo[t]; a[8 + t] = hi[t]; }
            afrag[kk] = a;
        }
    }

    // cooperative async panel fetch, fully uniform over 256 threads:
    //   bytes [0,4096)    : 256 x b128 chunks
    //   bytes [4096,6144) : 256 x b64 chunks
    const char* wmbase = (const char*)wmp;
    unsigned lds0 = (unsigned)(size_t)(&bpanel[0][0]);
    unsigned lds1 = (unsigned)(size_t)(&bpanel[1][0]);
    unsigned t16 = threadIdx.x * 16u;
    unsigned t8  = 4096u + threadIdx.x * 8u;
    auto fetch_panel = [&](int ct, int nb) {
        unsigned ldsb = (nb ? lds1 : lds0);
        const char* g = wmbase + (long long)ct * 6144;
        async_copy_b128(ldsb + t16, g + t16);
        async_copy_b64(ldsb + t8, g + t8);
    };

    fetch_panel(0, 0);
    wait_async0();
    __syncthreads();

    bool storer = active && (ln == 0);
    for (int i = 0; i < HID; ++i) {
        float s[8] = {0, 0, 0, 0, 0, 0, 0, 0};
#pragma unroll
        for (int sub = 0; sub < 4; ++sub) {      // ct = 4i+sub -> buffer = sub&1
            int ct = i * 4 + sub;
            if (sub < 3 || i < HID - 1) fetch_panel(ct + 1, (sub & 1) ^ 1);
            const __bf16* bp = &bpanel[sub & 1][lane * 16];
            v8f c = {};
#pragma unroll
            for (int kk = 0; kk < 6; ++kk) {
                v16bf b = *(const v16bf*)(bp + kk * 512);
                c = __builtin_amdgcn_wmma_f32_16x16x32_bf16(false, afrag[kk], false, b,
                                                            (short)0, c, false, false);
            }
            float bmv = bm[ct * 16 + ln];        // bias of this mm column
#pragma unroll
            for (int v = 0; v < 8; ++v)
                s[v] += (c[v] + bmv) * hrreg[sub][v];
            wait_async0();
            __syncthreads();
        }
        // reduce the 16 j-lanes within each half-wave (single ds_swizzle/step)
#pragma unroll
        for (int v = 0; v < 8; ++v) {
            float t = s[v];
            t = xor_add<1>(t);
            t = xor_add<2>(t);
            t = xor_add<4>(t);
            t = xor_add<8>(t);
            if (storer)                          // lanes 0 and 16: 8 rows each
                hid1[(long long)(row0 + mboff + v) * HID + i] = t;
        }
    }
}

// hid = relu(hid1 + bbias + relu(res))
__global__ void final_kernel(const float* hid1, const float* bbias,
                             const float* res, float* out, long long n64) {
    long long i = (long long)blockIdx.x * blockDim.x + threadIdx.x;
    if (i >= n64) return;
    float hr = res[i] > 0.0f ? res[i] : 0.0f;
    float v = hid1[i] + bbias[i] + hr;
    out[i] = v > 0.0f ? v : 0.0f;
}

// ------------------------------- launcher ----------------------------------

extern "C" void kernel_launch(void* const* d_in, const int* in_sizes, int n_in,
                              void* d_out, int out_size, void* d_ws, size_t ws_size,
                              hipStream_t stream) {
    const float*      x  = (const float*)d_in[0];
    const long long*  ei = (const long long*)d_in[1];   // int64 edge_index [2,E]
    const float* W1 = (const float*)d_in[2];
    const float* b1 = (const float*)d_in[3];
    const float* W2 = (const float*)d_in[4];
    const float* b2 = (const float*)d_in[5];
    const float* Wd = (const float*)d_in[6];
    const float* bd = (const float*)d_in[7];
    const float* Wm = (const float*)d_in[8];
    const float* bm = (const float*)d_in[9];
    const float* Wb = (const float*)d_in[10];
    const float* bb = (const float*)d_in[11];

    const int N = in_sizes[0] / NFEAT;
    const int E = in_sizes[1] / 2;

    float* res_out = (float*)d_out;               // [N,64]
    float* hid_out = (float*)d_out + (long long)N * HID;

    // workspace carve-out (256B aligned)
    char* p = (char*)d_ws;
    auto alloc = [&](size_t bytes) -> void* {
        void* r = (void*)p; p += (bytes + 255) & ~(size_t)255; return r;
    };
    float*  dinv  = (float*)alloc((size_t)N * 4);
    __bf16* x_bf  = (__bf16*)alloc((size_t)N * NFEAT * 2);
    __bf16* z_bf  = (__bf16*)alloc((size_t)N * HID3 * 2);
    __bf16* h_bf  = (__bf16*)alloc((size_t)N * HID2 * 2);
    float*  hw1   = (float*)alloc((size_t)N * HID2 * 4);
    float*  agg1  = (float*)alloc((size_t)N * HID2 * 4);
    float*  hw2   = (float*)alloc((size_t)N * HID * 4);
    float*  agg2  = (float*)alloc((size_t)N * HID * 4);
    float*  bbias = (float*)alloc((size_t)N * HID * 4);
    float*  hid1  = (float*)alloc((size_t)N * HID * 4);
    __bf16* W1p   = (__bf16*)alloc((size_t)NFEAT * HID2 * 2);
    __bf16* Wdp   = (__bf16*)alloc((size_t)NFEAT * HID2 * 2);
    __bf16* W2p   = (__bf16*)alloc((size_t)HID2 * HID * 2);
    __bf16* Wbp   = (__bf16*)alloc((size_t)HID3 * HID * 2);
    __bf16* Wmp   = (__bf16*)alloc((size_t)HID3 * WMCOL * 2);
    (void)ws_size; (void)n_in; (void)out_size;

    auto cdiv = [](long long a, long long b) { return (unsigned)((a + b - 1) / b); };

    // degree / dinv
    init_deg<<<cdiv(N, 256), 256, 0, stream>>>(dinv, N);
    accum_deg<<<cdiv(E, 256), 256, 0, stream>>>(ei, dinv, E);
    make_dinv<<<cdiv(N, 256), 256, 0, stream>>>(dinv, N);

    // conversions + weight packing
    cvt_f32_bf16<<<cdiv((long long)N * NFEAT, 256), 256, 0, stream>>>(x, x_bf, (long long)N * NFEAT);
    pack_b<<<cdiv((HID2 / 16) * (NFEAT / 32) * 32, 256), 256, 0, stream>>>(W1, W1p, NFEAT, HID2);
    pack_b<<<cdiv((HID2 / 16) * (NFEAT / 32) * 32, 256), 256, 0, stream>>>(Wd, Wdp, NFEAT, HID2);
    pack_b<<<cdiv((HID / 16) * (HID2 / 32) * 32, 256), 256, 0, stream>>>(W2, W2p, HID2, HID);
    pack_b<<<cdiv((HID / 16) * (HID3 / 32) * 32, 256), 256, 0, stream>>>(Wb, Wbp, HID3, HID);
    pack_b<<<cdiv((WMCOL / 16) * (HID3 / 32) * 32, 256), 256, 0, stream>>>(Wm, Wmp, HID3, WMCOL);

    // hw1 = x@W1 ; xd = x@Wd + bd -> z[:,0:128] (bf16)
    {
        unsigned tiles = (N / 16) * (HID2 / 16);
        wmma_gemm<NFEAT / 32><<<cdiv(tiles, 4), 128, 0, stream>>>(x_bf, W1p, nullptr,
            hw1, nullptr, HID2, N, HID2);
        wmma_gemm<NFEAT / 32><<<cdiv(tiles, 4), 128, 0, stream>>>(x_bf, Wdp, bd,
            nullptr, z_bf, HID3, N, HID2);
    }

    // layer 1 aggregate
    zero_f32<<<cdiv((long long)N * HID2, 256), 256, 0, stream>>>(agg1, (long long)N * HID2);
    scatter_edges<<<cdiv(E, 4), 128, 0, stream>>>(ei, dinv, hw1, agg1, E, HID2);
    finish1<<<cdiv((long long)N * HID2, 256), 256, 0, stream>>>(agg1, hw1, dinv, b1, h_bf, N);

    // layer 2
    {
        unsigned tiles = (N / 16) * (HID / 16);
        wmma_gemm<HID2 / 32><<<cdiv(tiles, 4), 128, 0, stream>>>(h_bf, W2p, nullptr,
            hw2, nullptr, HID, N, HID);
    }
    zero_f32<<<cdiv((long long)N * HID, 256), 256, 0, stream>>>(agg2, (long long)N * HID);
    scatter_edges<<<cdiv(E, 4), 128, 0, stream>>>(ei, dinv, hw2, agg2, E, HID);
    finish2<<<cdiv((long long)N * HID, 256), 256, 0, stream>>>(agg2, hw2, dinv, b2, res_out, z_bf, N);

    // bbias = z@Wb + bb
    {
        unsigned tiles = (N / 16) * (HID / 16);
        wmma_gemm<HID3 / 32><<<cdiv(tiles, 4), 128, 0, stream>>>(z_bf, Wbp, bb,
            bbias, nullptr, HID, N, HID);
    }

    // fused hypernetwork: hid1 = (z@Wm+bm) · relu(res)
    hyper_kernel<<<cdiv(N / 16, 8), 256, 0, stream>>>(z_bf, Wmp, res_out, bm, hid1, N);

    // hid = relu(hid1 + bbias + relu(res))
    final_kernel<<<cdiv((long long)N * HID, 256), 256, 0, stream>>>(hid1, bbias, res_out, hid_out,
                                                                    (long long)N * HID);
}
